// GraphAttentionEmbedding_78941498901092
// MI455X (gfx1250) — matrix operand order repaired and checked by reading, not compiled, not run on hardware
//
#include <hip/hip_runtime.h>
#include <math.h>

#define N_NODES 100000
#define N_EDGES 640000
#define CH      128
#define HEADS   4
#define HEAD_C  32
#define TPW     10          // row tiles per wave; 6250 % 10 == 0 and 40000 % 10 == 0

typedef __attribute__((ext_vector_type(16))) __bf16 v16bf;
typedef __attribute__((ext_vector_type(8)))  float  v8f;

union Frag32 { int4 i4[2]; v16bf v; };

__device__ __forceinline__ unsigned short f2bf(float f) {
  unsigned int u = __float_as_uint(f);
  u += 0x7FFFu + ((u >> 16) & 1u);          // round-to-nearest-even
  return (unsigned short)(u >> 16);
}
__device__ __forceinline__ float bf2f(unsigned short h) {
  return __uint_as_float(((unsigned int)h) << 16);
}
__device__ __forceinline__ void atomicMaxF(float* addr, float val) {
  if (val >= 0.0f) atomicMax((int*)addr, __float_as_int(val));
  else             atomicMin((unsigned int*)addr, __float_as_uint(val));
}

__device__ __forceinline__ void load_a4(Frag32* A, const unsigned short* xrow, int kh) {
#pragma unroll
  for (int j = 0; j < 4; ++j) {
    A[j].i4[0] = *(const int4*)(xrow + j * 32 + 8 * kh);
    A[j].i4[1] = *(const int4*)(xrow + j * 32 + 16 + 8 * kh);
  }
}

// ---- weight convert: [128(in),128(out)] f32 -> transposed [out,in] bf16 ----
__global__ void k_wconv(const float* __restrict__ W, unsigned short* __restrict__ WT) {
  int idx = blockIdx.x * 256 + threadIdx.x;
  if (idx >= CH * CH) return;
  int i = idx >> 7, o = idx & 127;
  WT[o * CH + i] = f2bf(W[i * CH + o]);
}

// ---- f32 -> bf16 vector convert ----
__global__ void k_f2b(const float* __restrict__ src, unsigned short* __restrict__ dst, int n) {
  int idx = blockIdx.x * 256 + threadIdx.x;
  if (idx < n) dst[idx] = f2bf(src[idx]);
}

// ---- edge_attr = [cos(rel_t*Wt+bt) || msg] in bf16, [E,128] ----
__global__ void k_edge_attr(const float* __restrict__ last_update, const float* __restrict__ t,
                            const float* __restrict__ msg, const int* __restrict__ src,
                            const float* __restrict__ Wt, const float* __restrict__ bt,
                            unsigned short* __restrict__ eattr) {
  int idx = blockIdx.x * 256 + threadIdx.x;
  if (idx >= N_EDGES * CH) return;
  int e = idx >> 7, col = idx & 127;
  float v;
  if (col < 64) {
    float rt = last_update[src[e]] - t[e];
    v = cosf(rt * Wt[col] + bt[col]);
  } else {
    v = msg[e * 64 + (col - 64)];
  }
  eattr[idx] = f2bf(v);
}

// ---- B-resident, statically pipelined GEMM, f32 out + bias: Y = Xb @ W + b ----
__global__ __launch_bounds__(128)
void k_gemm_f32(const unsigned short* __restrict__ Xb, const unsigned short* __restrict__ W,
                const float* __restrict__ bias, float* __restrict__ Y, int rowTiles) {
  const int gw   = (int)((blockIdx.x * blockDim.x + threadIdx.x) >> 5);
  const int lane = threadIdx.x & 31;
  const int ct   = gw & 7;
  const int rt0  = (gw >> 3) * TPW;
  if (rt0 >= rowTiles) return;                      // wave-uniform
  const int rlo = lane & 15;
  const int kh  = lane >> 4;
  // resident B fragments for this 16-column slice (all 4 K-chunks)
  Frag32 B[4];
  const int wrow = (ct * 16 + rlo) * CH;
#pragma unroll
  for (int j = 0; j < 4; ++j) {
    const int kb = j * 32 + 8 * kh;
    B[j].i4[0] = *(const int4*)(W + wrow + kb);
    B[j].i4[1] = *(const int4*)(W + wrow + kb + 16);
  }
  const int col = ct * 16 + rlo;
  const float bb = bias[col];
  const unsigned short* xbase = Xb + (size_t)(rt0 * 16 + rlo) * CH;
  Frag32 A0[4], A1[4];
  load_a4(A0, xbase, kh);
#pragma unroll
  for (int i = 0; i < TPW; ++i) {
    Frag32* Acur = (i & 1) ? A1 : A0;   // compile-time after full unroll
    Frag32* Anxt = (i & 1) ? A0 : A1;
    if (i + 1 < TPW) load_a4(Anxt, xbase + (size_t)(i + 1) * 16 * CH, kh);
    v8f acc = {};
#pragma unroll
    for (int j = 0; j < 4; ++j)
      acc = __builtin_amdgcn_wmma_f32_16x16x32_bf16(false, Acur[j].v, false, B[j].v,
                                                    (short)0, acc, false, false);
#pragma unroll
    for (int g = 0; g < 8; ++g)
      Y[(size_t)((rt0 + i) * 16 + g + 8 * kh) * CH + col] = acc[g] + bb;
  }
}

// ---- B-resident, statically pipelined GEMM, bf16 out, no bias: e = edge_attr @ We ----
__global__ __launch_bounds__(128)
void k_gemm_b16(const unsigned short* __restrict__ Xb, const unsigned short* __restrict__ W,
                unsigned short* __restrict__ Yb, int rowTiles) {
  const int gw   = (int)((blockIdx.x * blockDim.x + threadIdx.x) >> 5);
  const int lane = threadIdx.x & 31;
  const int ct   = gw & 7;
  const int rt0  = (gw >> 3) * TPW;
  if (rt0 >= rowTiles) return;
  const int rlo = lane & 15;
  const int kh  = lane >> 4;
  Frag32 B[4];
  const int wrow = (ct * 16 + rlo) * CH;
#pragma unroll
  for (int j = 0; j < 4; ++j) {
    const int kb = j * 32 + 8 * kh;
    B[j].i4[0] = *(const int4*)(W + wrow + kb);
    B[j].i4[1] = *(const int4*)(W + wrow + kb + 16);
  }
  const int col = ct * 16 + rlo;
  const unsigned short* xbase = Xb + (size_t)(rt0 * 16 + rlo) * CH;
  Frag32 A0[4], A1[4];
  load_a4(A0, xbase, kh);
#pragma unroll
  for (int i = 0; i < TPW; ++i) {
    Frag32* Acur = (i & 1) ? A1 : A0;
    Frag32* Anxt = (i & 1) ? A0 : A1;
    if (i + 1 < TPW) load_a4(Anxt, xbase + (size_t)(i + 1) * 16 * CH, kh);
    v8f acc = {};
#pragma unroll
    for (int j = 0; j < 4; ++j)
      acc = __builtin_amdgcn_wmma_f32_16x16x32_bf16(false, Acur[j].v, false, B[j].v,
                                                    (short)0, acc, false, false);
#pragma unroll
    for (int g = 0; g < 8; ++g)
      Yb[(size_t)((rt0 + i) * 16 + g + 8 * kh) * CH + col] = f2bf(acc[g]);
  }
}

// ---- init segment-softmax state ----
__global__ void k_init(float* __restrict__ amax, float* __restrict__ denom, int n) {
  int idx = blockIdx.x * 256 + threadIdx.x;
  if (idx < n) { amax[idx] = -1e30f; denom[idx] = 0.0f; }
}

// ---- attention logits + segment max ----
__global__ void k_alpha(const float* __restrict__ Q, const float* __restrict__ K,
                        const unsigned short* __restrict__ EB,
                        const int* __restrict__ src, const int* __restrict__ dst,
                        float* __restrict__ alpha, float* __restrict__ amax) {
  int idx = blockIdx.x * 256 + threadIdx.x;
  if (idx >= N_EDGES * HEADS) return;
  int e = idx >> 2, h = idx & 3;
  int s = src[e], d = dst[e];
  const float* q = Q + (size_t)d * CH + h * HEAD_C;
  const float* k = K + (size_t)s * CH + h * HEAD_C;
  const unsigned short* eb = EB + (size_t)e * CH + h * HEAD_C;
  float acc = 0.0f;
#pragma unroll
  for (int c = 0; c < HEAD_C; c += 4) {
    float4  qv = *(const float4*)(q + c);
    float4  kv = *(const float4*)(k + c);
    ushort4 ev = *(const ushort4*)(eb + c);
    acc += qv.x * (kv.x + bf2f(ev.x));
    acc += qv.y * (kv.y + bf2f(ev.y));
    acc += qv.z * (kv.z + bf2f(ev.z));
    acc += qv.w * (kv.w + bf2f(ev.w));
  }
  acc *= 0.17677669529663687f;  // 1/sqrt(32)
  alpha[idx] = acc;
  atomicMaxF(amax + (size_t)d * HEADS + h, acc);
}

// ---- exp(alpha - max) + segment sum ----
__global__ void k_expa(float* __restrict__ alpha, const float* __restrict__ amax,
                       float* __restrict__ denom, const int* __restrict__ dst) {
  int idx = blockIdx.x * 256 + threadIdx.x;
  if (idx >= N_EDGES * HEADS) return;
  int e = idx >> 2, h = idx & 3;
  int d = dst[e];
  float ex = expf(alpha[idx] - amax[(size_t)d * HEADS + h]);
  alpha[idx] = ex;
  atomicAdd(denom + (size_t)d * HEADS + h, ex);
}

// ---- weighted scatter-aggregate: OUT[dst] += attn * (v[src] + e); 4 cols/thread ----
__global__ void k_agg(const float* __restrict__ alpha, const float* __restrict__ denom,
                      const float* __restrict__ V, const unsigned short* __restrict__ EB,
                      const int* __restrict__ src, const int* __restrict__ dst,
                      float* __restrict__ OUT) {
  int idx = blockIdx.x * 256 + threadIdx.x;
  if (idx >= N_EDGES * (CH / 4)) return;
  int e = idx >> 5, col = (idx & 31) * 4, h = col >> 5;
  int s = src[e], d = dst[e];
  float attn = alpha[(size_t)e * HEADS + h] / (denom[(size_t)d * HEADS + h] + 1e-16f);
  float4  vv = *(const float4*)(V + (size_t)s * CH + col);
  ushort4 ev = *(const ushort4*)(EB + (size_t)e * CH + col);
  float* o = OUT + (size_t)d * CH + col;
  atomicAdd(o + 0, attn * (vv.x + bf2f(ev.x)));
  atomicAdd(o + 1, attn * (vv.y + bf2f(ev.y)));
  atomicAdd(o + 2, attn * (vv.z + bf2f(ev.z)));
  atomicAdd(o + 3, attn * (vv.w + bf2f(ev.w)));
}

extern "C" void kernel_launch(void* const* d_in, const int* in_sizes, int n_in,
                              void* d_out, int out_size, void* d_ws, size_t ws_size,
                              hipStream_t stream) {
  const float* x    = (const float*)d_in[0];
  const float* lupd = (const float*)d_in[1];
  const float* t    = (const float*)d_in[2];
  const float* msg  = (const float*)d_in[3];
  const int*   eidx = (const int*)d_in[4];
  const float* Wt   = (const float*)d_in[5];
  const float* bt   = (const float*)d_in[6];
  const float* Wf[10] = { (const float*)d_in[7],  (const float*)d_in[9],
                          (const float*)d_in[11], (const float*)d_in[14],
                          (const float*)d_in[13],                         // layer1: q,k,v,skip,edge
                          (const float*)d_in[16], (const float*)d_in[18],
                          (const float*)d_in[20], (const float*)d_in[23],
                          (const float*)d_in[22] };                       // layer2: q,k,v,skip,edge
  const float* bq1 = (const float*)d_in[8],  *bk1 = (const float*)d_in[10];
  const float* bv1 = (const float*)d_in[12], *bs1 = (const float*)d_in[15];
  const float* bq2 = (const float*)d_in[17], *bk2 = (const float*)d_in[19];
  const float* bv2 = (const float*)d_in[21], *bs2 = (const float*)d_in[24];
  const int* srcI = eidx;
  const int* dstI = eidx + N_EDGES;
  float* out = (float*)d_out;

  // workspace carve (256B aligned)
  char* p = (char*)d_ws;
  auto carve = [&](size_t bytes) { void* r = (void*)p; p += (bytes + 255) & ~(size_t)255; return r; };
  unsigned short* WB    = (unsigned short*)carve(10 * CH * CH * sizeof(unsigned short));
  unsigned short* XB    = (unsigned short*)carve((size_t)N_NODES * CH * sizeof(unsigned short));
  unsigned short* EATTR = (unsigned short*)carve((size_t)N_EDGES * CH * sizeof(unsigned short));
  unsigned short* EB    = (unsigned short*)carve((size_t)N_EDGES * CH * sizeof(unsigned short));
  float* Q     = (float*)carve((size_t)N_NODES * CH * sizeof(float));
  float* K     = (float*)carve((size_t)N_NODES * CH * sizeof(float));
  float* V     = (float*)carve((size_t)N_NODES * CH * sizeof(float));
  float* H     = (float*)carve((size_t)N_NODES * CH * sizeof(float));
  float* ALPHA = (float*)carve((size_t)N_EDGES * HEADS * sizeof(float));
  float* AMAX  = (float*)carve((size_t)N_NODES * HEADS * sizeof(float));
  float* DENOM = (float*)carve((size_t)N_NODES * HEADS * sizeof(float));

  const int nodeTiles = N_NODES / 16;               // 6250 (= 625 * TPW)
  const int edgeTiles = N_EDGES / 16;               // 40000 (= 4000 * TPW)
  const int gW   = (CH * CH + 255) / 256;
  const int gXN  = (N_NODES * CH + 255) / 256;
  const int gEA  = (N_EDGES * CH + 255) / 256;
  const int gEH  = (N_EDGES * HEADS + 255) / 256;
  const int gNI  = (N_NODES * HEADS + 255) / 256;
  const int gAG  = (N_EDGES * (CH / 4) + 255) / 256;
  const int gGN  = ((nodeTiles / TPW) * 8 * 32 + 127) / 128;  // node GEMM blocks
  const int gGE  = ((edgeTiles / TPW) * 8 * 32 + 127) / 128;  // edge GEMM blocks

  // weights -> transposed bf16
  for (int i = 0; i < 10; ++i)
    k_wconv<<<gW, 256, 0, stream>>>(Wf[i], WB + (size_t)i * CH * CH);
  // edge attributes (shared by both layers)
  k_edge_attr<<<gEA, 256, 0, stream>>>(lupd, t, msg, srcI, Wt, bt, EATTR);
  // x -> bf16
  k_f2b<<<gXN, 256, 0, stream>>>(x, XB, N_NODES * CH);

  // -------- layer 1 --------
  k_gemm_f32<<<gGN, 128, 0, stream>>>(XB, WB + 0 * CH * CH, bq1, Q, nodeTiles);
  k_gemm_f32<<<gGN, 128, 0, stream>>>(XB, WB + 1 * CH * CH, bk1, K, nodeTiles);
  k_gemm_f32<<<gGN, 128, 0, stream>>>(XB, WB + 2 * CH * CH, bv1, V, nodeTiles);
  k_gemm_f32<<<gGN, 128, 0, stream>>>(XB, WB + 3 * CH * CH, bs1, H, nodeTiles);
  k_gemm_b16<<<gGE, 128, 0, stream>>>(EATTR, WB + 4 * CH * CH, EB, edgeTiles);
  k_init<<<gNI, 256, 0, stream>>>(AMAX, DENOM, N_NODES * HEADS);
  k_alpha<<<gEH, 256, 0, stream>>>(Q, K, EB, srcI, dstI, ALPHA, AMAX);
  k_expa<<<gEH, 256, 0, stream>>>(ALPHA, AMAX, DENOM, dstI);
  k_agg<<<gAG, 256, 0, stream>>>(ALPHA, DENOM, V, EB, srcI, dstI, H);

  // -------- layer 2 --------
  k_f2b<<<gXN, 256, 0, stream>>>(H, XB, N_NODES * CH);
  k_gemm_f32<<<gGN, 128, 0, stream>>>(XB, WB + 5 * CH * CH, bq2, Q, nodeTiles);
  k_gemm_f32<<<gGN, 128, 0, stream>>>(XB, WB + 6 * CH * CH, bk2, K, nodeTiles);
  k_gemm_f32<<<gGN, 128, 0, stream>>>(XB, WB + 7 * CH * CH, bv2, V, nodeTiles);
  k_gemm_f32<<<gGN, 128, 0, stream>>>(XB, WB + 8 * CH * CH, bs2, out, nodeTiles);
  k_gemm_b16<<<gGE, 128, 0, stream>>>(EATTR, WB + 9 * CH * CH, EB, edgeTiles);
  k_init<<<gNI, 256, 0, stream>>>(AMAX, DENOM, N_NODES * HEADS);
  k_alpha<<<gEH, 256, 0, stream>>>(Q, K, EB, srcI, dstI, ALPHA, AMAX);
  k_expa<<<gEH, 256, 0, stream>>>(ALPHA, AMAX, DENOM, dstI);
  k_agg<<<gAG, 256, 0, stream>>>(ALPHA, DENOM, V, EB, srcI, dstI, out);

  (void)in_sizes; (void)n_in; (void)out_size; (void)ws_size;
}
